// StabilizedNeuralODE_24687472017939
// MI455X (gfx1250) — compile-verified, matrix operand order checked by reading, not run on hardware
//
#include <hip/hip_runtime.h>
#include <hip/hip_bf16.h>

typedef __attribute__((ext_vector_type(16))) _Float16 v16h;
typedef __attribute__((ext_vector_type(8)))  _Float16 v8h;
typedef __attribute__((ext_vector_type(8)))  float    v8f;

#define DD 128
#define WW 512
#define BB 512
#define TT 512

// RK tableau (rows: input coeffs for stage s; BCF: final combine)
__constant__ float ACF[6][5] = {
  {0.f, 0.f, 0.f, 0.f, 0.f},
  {0.161f, 0.f, 0.f, 0.f, 0.f},
  {-0.008480655492356989f, 0.335480655492357f, 0.f, 0.f, 0.f},
  {2.8971530571054935f, -6.359448489975075f, 4.3622954328695815f, 0.f, 0.f},
  {5.325864828439257f, -11.748883564062828f, 7.4955393428898365f, -0.09249506636175525f, 0.f},
  {5.86145544294642f, -12.92096931784711f, 8.159367898576159f, -0.071584973281401f, -0.028269050394068383f}
};
__constant__ float BCF[6] = {
  0.09646076681806523f, 0.01f, 0.4798896504144996f,
  1.379008574103742f, -3.290069515436081f, 2.324710524099774f
};

// Fast tanh via hardware transcendentals: tanh(x) = 1 - 2/(exp(2x)+1).
__device__ __forceinline__ float fast_tanh(float x) {
  float e = __builtin_amdgcn_exp2f(x * 2.8853900817779268f); // 2/ln(2)
  float r = __builtin_amdgcn_rcpf(e + 1.0f);
  return __builtin_fmaf(-2.0f, r, 1.0f);
}

// Pack fp32 row-major weight (N rows, K cols; used as B = w^T, K x N) into
// per-fragment f16 layout so each lane's WMMA B-operand is one contiguous
// 32-byte load. Fragment = 16(N) x 32(K).
__global__ void pack_b_kernel(const float* __restrict__ w, _Float16* __restrict__ dst,
                              int N, int K) {
  int tid = blockIdx.x * blockDim.x + threadIdx.x;
  int KT = K >> 5;
  int nthr = (N >> 4) * KT * 32;
  if (tid >= nthr) return;
  int frag = tid >> 5;
  int lane = tid & 31;
  int n_t = frag / KT;
  int k_t = frag - n_t * KT;
  int n = n_t * 16 + (lane & 15);
  int kbase = k_t * 32 + ((lane >> 4) << 3);
  union { v16h v; _Float16 e[16]; } u;
#pragma unroll
  for (int j = 0; j < 8; ++j) {
    u.e[j]     = (_Float16)w[(size_t)n * K + kbase + j];
    u.e[j + 8] = (_Float16)w[(size_t)n * K + kbase + 16 + j];
  }
  *(v16h*)(dst + (size_t)frag * 512 + lane * 16) = u.v;
}

// Gather one 16x32 f16 A-fragment from an LDS row-major (16 x stride) tile.
__device__ __forceinline__ v16h load_afrag(const _Float16* src, int stride, int kt, int lane) {
  int m  = lane & 15;
  int st = kt * 32 + ((lane >> 4) << 3);
  const _Float16* p = src + m * stride + st;
  union { v16h v; v8h h[2]; } u;
  u.h[0] = *(const v8h*)(p);
  u.h[1] = *(const v8h*)(p + 16);
  return u.v;
}

__device__ __forceinline__ v8f wmma16(v16h a, v16h b, v8f c) {
  return __builtin_amdgcn_wmma_f32_16x16x32_f16(false, a, false, b, (short)0, c, false, false);
}

// Launder a zero offset through an empty asm. Adding it to the weight base
// pointers makes every fragment address non-loop-invariant (stops LICM from
// hoisting all weight loads out of the time loop -> no scratch spills), while
// keeping the pointer chain rooted at the kernel argument so the loads still
// infer the GLOBAL address space (global_load_b128, LOADcnt only) instead of
// decaying to flat_load (which would also tie up DScnt).
__device__ __forceinline__ size_t launder_off() {
  size_t off = 0;
  asm volatile("" : "+s"(off));
  return off;
}

// One workgroup integrates 16 independent batch rows for all T steps.
// 256 threads = 8 wave32. No inter-workgroup communication needed.
// Weights are streamed from L2 (they stay hot: 0.8 MB shared by all WGs);
// activations are staged in LDS; WMMA accumulates in fp32.
__global__ void __launch_bounds__(256)
ode_kernel(const float* __restrict__ ts, const float* __restrict__ yi,
           const _Float16* __restrict__ w1p, const float* __restrict__ b1,
           const _Float16* __restrict__ w2p, const float* __restrict__ b2,
           const _Float16* __restrict__ w3p, const float* __restrict__ b3,
           const _Float16* __restrict__ Ap, float* __restrict__ out)
{
  extern __shared__ char smem[];
  float*    yS  = (float*)smem;                      //  8 KB : y (16 x 128 fp32)
  float*    kS  = (float*)(smem + 8 * 1024);         // 48 KB : k1..k6 (fp32)
  _Float16* yhS = (_Float16*)(smem + 56 * 1024);     //  4 KB : stage input f16
  _Float16* h1S = (_Float16*)(smem + 60 * 1024);     // 16 KB : layer1 act f16
  _Float16* h2S = (_Float16*)(smem + 76 * 1024);     // 16 KB : layer2 act f16
                                                     // total 92 KB

  const int tid  = threadIdx.x;
  const int lane = tid & 31;
  const int wave = tid >> 5;
  const int m0   = blockIdx.x * 16;

  // Load initial y; emit t=0 output.
  for (int i = tid; i < 16 * DD; i += 256) {
    int m = i >> 7, d = i & (DD - 1);
    float v = yi[(size_t)(m0 + m) * DD + d];
    yS[i] = v;
    out[((size_t)(m0 + m) * TT + 0) * DD + d] = v;
  }
  __syncthreads();

#pragma unroll 1
  for (int t = 0; t < TT - 1; ++t) {
    float dt = ts[t + 1] - ts[t];

#pragma unroll 1
    for (int s = 0; s < 6; ++s) {
      // Fresh (non-invariant, but still provably-global) weight pointers.
      size_t lo = launder_off();
      const _Float16* w1l = w1p + lo;
      const _Float16* w2l = w2p + lo;
      const _Float16* w3l = w3p + lo;
      const _Float16* Al  = Ap + lo;

      // Stage input: yh = f16(y + dt * sum_j a[s][j] * k_j)
      for (int i = tid; i < 16 * DD; i += 256) {
        float acc = yS[i];
#pragma unroll
        for (int j = 0; j < 5; ++j)
          if (j < s) acc += dt * ACF[s][j] * kS[j * 16 * DD + i];
        yhS[i] = (_Float16)acc;
      }
      __syncthreads();

      // ---- Layer 1: (16xD) @ (DxW) -> tanh -> h1 (16xW). 4 N-tiles/wave.
      {
        v8f acc[4] = {};
        const int KT = DD / 32; // 4
#pragma unroll
        for (int kt = 0; kt < KT; ++kt) {
          v16h a = load_afrag(yhS, DD, kt, lane);
#pragma unroll
          for (int tt = 0; tt < 4; ++tt) {
            int frag = (wave * 4 + tt) * KT + kt;
            v16h b = *(const v16h*)(w1l + (size_t)frag * 512 + lane * 16);
            acc[tt] = wmma16(a, b, acc[tt]);
          }
        }
#pragma unroll
        for (int tt = 0; tt < 4; ++tt) {
          int n = (wave * 4 + tt) * 16 + (lane & 15);
          float bias = b1[n];
          int mb = (lane >> 4) << 3;
#pragma unroll
          for (int r = 0; r < 8; ++r)
            h1S[(mb + r) * WW + n] = (_Float16)fast_tanh(acc[tt][r] + bias);
        }
      }
      __syncthreads();

      // ---- Layer 2: (16xW) @ (WxW) -> tanh -> h2 (16xW). 4 N-tiles/wave.
      {
        v8f acc[4] = {};
        const int KT = WW / 32; // 16
#pragma unroll
        for (int kt = 0; kt < KT; ++kt) {
          v16h a = load_afrag(h1S, WW, kt, lane);
#pragma unroll
          for (int tt = 0; tt < 4; ++tt) {
            int frag = (wave * 4 + tt) * KT + kt;
            v16h b = *(const v16h*)(w2l + (size_t)frag * 512 + lane * 16);
            acc[tt] = wmma16(a, b, acc[tt]);
          }
        }
#pragma unroll
        for (int tt = 0; tt < 4; ++tt) {
          int n = (wave * 4 + tt) * 16 + (lane & 15);
          float bias = b2[n];
          int mb = (lane >> 4) << 3;
#pragma unroll
          for (int r = 0; r < 8; ++r)
            h2S[(mb + r) * WW + n] = (_Float16)fast_tanh(acc[tt][r] + bias);
        }
      }
      __syncthreads();

      // ---- Layer 3 + stabilizer: h2 @ w3^T + y @ A^T + b3 -> k_s (fp32).
      // One 16-col tile per wave; two independent accumulation chains.
      {
        v8f acc0 = {}, acc1 = {};
        const int KT3 = WW / 32; // 16
#pragma unroll
        for (int kt = 0; kt < KT3; kt += 2) {
          v16h a0 = load_afrag(h2S, WW, kt, lane);
          v16h a1 = load_afrag(h2S, WW, kt + 1, lane);
          v16h b0 = *(const v16h*)(w3l + (size_t)(wave * KT3 + kt) * 512 + lane * 16);
          v16h b1v = *(const v16h*)(w3l + (size_t)(wave * KT3 + kt + 1) * 512 + lane * 16);
          acc0 = wmma16(a0, b0, acc0);
          acc1 = wmma16(a1, b1v, acc1);
        }
        const int KTA = DD / 32; // 4
#pragma unroll
        for (int kt = 0; kt < KTA; kt += 2) {
          v16h a0 = load_afrag(yhS, DD, kt, lane);
          v16h a1 = load_afrag(yhS, DD, kt + 1, lane);
          v16h b0 = *(const v16h*)(Al + (size_t)(wave * KTA + kt) * 512 + lane * 16);
          v16h b1v = *(const v16h*)(Al + (size_t)(wave * KTA + kt + 1) * 512 + lane * 16);
          acc0 = wmma16(a0, b0, acc0);
          acc1 = wmma16(a1, b1v, acc1);
        }
        int n = wave * 16 + (lane & 15);
        float bias = b3[n];
        int mb = (lane >> 4) << 3;
        float* kdst = kS + s * 16 * DD;
#pragma unroll
        for (int r = 0; r < 8; ++r)
          kdst[(mb + r) * DD + n] = acc0[r] + acc1[r] + bias;
      }
      __syncthreads();
    }

    // Final combine: y += dt * sum_j B_j * k_j ; emit output row t+1.
    for (int i = tid; i < 16 * DD; i += 256) {
      float v = yS[i];
#pragma unroll
      for (int j = 0; j < 6; ++j) v += dt * BCF[j] * kS[j * 16 * DD + i];
      yS[i] = v;
      int m = i >> 7, d = i & (DD - 1);
      out[((size_t)(m0 + m) * TT + (t + 1)) * DD + d] = v;
    }
    __syncthreads();
  }
}

extern "C" void kernel_launch(void* const* d_in, const int* in_sizes, int n_in,
                              void* d_out, int out_size, void* d_ws, size_t ws_size,
                              hipStream_t stream) {
  const float* ts = (const float*)d_in[0];
  const float* yi = (const float*)d_in[1];
  const float* w1 = (const float*)d_in[2];
  const float* b1 = (const float*)d_in[3];
  const float* w2 = (const float*)d_in[4];
  const float* b2 = (const float*)d_in[5];
  const float* w3 = (const float*)d_in[6];
  const float* b3 = (const float*)d_in[7];
  const float* A  = (const float*)d_in[8];

  _Float16* w1p = (_Float16*)d_ws;           // 512x128 -> 65536 halves
  _Float16* w2p = w1p + 512 * 128;           // 512x512 -> 262144 halves
  _Float16* w3p = w2p + 512 * 512;           // 128x512 -> 65536 halves
  _Float16* Ap  = w3p + 128 * 512;           // 128x128 -> 16384 halves

  // One-time weight f16 conversion + WMMA B-fragment packing (L2-resident).
  pack_b_kernel<<<(32 * 4  * 32 + 255) / 256, 256, 0, stream>>>(w1, w1p, 512, 128);
  pack_b_kernel<<<(32 * 16 * 32 + 255) / 256, 256, 0, stream>>>(w2, w2p, 512, 512);
  pack_b_kernel<<<(8  * 16 * 32 + 255) / 256, 256, 0, stream>>>(w3, w3p, 128, 512);
  pack_b_kernel<<<(8  * 4  * 32 + 255) / 256, 256, 0, stream>>>(A,  Ap,  128, 128);

  size_t shmem = 92 * 1024;
  ode_kernel<<<BB / 16, 256, shmem, stream>>>(ts, yi, w1p, b1, w2p, b2, w3p, b3,
                                              Ap, (float*)d_out);
}
